// MultiQueryAttention_446676599437
// MI455X (gfx1250) — matrix operand order repaired and checked by reading, not compiled, not run on hardware
//
#include <hip/hip_runtime.h>

// ---------------------------------------------------------------------------
// MQA on gfx1250: bf16 WMMA GEMMs (TDM/LDS-tiled for the big ones) +
// flash-attention (online softmax), f32 accumulation.
// ---------------------------------------------------------------------------

typedef __attribute__((ext_vector_type(16))) __bf16 v16bf;
typedef __attribute__((ext_vector_type(8)))  float  v8f;
typedef __attribute__((ext_vector_type(4)))  unsigned int u32x4;
typedef __attribute__((ext_vector_type(8)))  int          i32x8;
typedef __attribute__((ext_vector_type(4)))  int          i32x4;

#define DEV __device__ __forceinline__

union FragU { uint4 u[2]; v16bf v; };

// Load a 16-element bf16 fragment as two 16-byte chunks (global or LDS).
DEV v16bf load_frag(const __bf16* p0, const __bf16* p1) {
  FragU f;
  f.u[0] = *(const uint4*)p0;
  f.u[1] = *(const uint4*)p1;
  return f.v;
}

DEV v8f wmma_bf16(v16bf a, v16bf b, v8f c) {
  return __builtin_amdgcn_wmma_f32_16x16x32_bf16(
      /*neg_a=*/false, a, /*neg_b=*/false, b,
      /*c_mod=*/(short)0, c, /*reuse_a=*/false, /*reuse_b=*/false);
}

// ---------------------------------------------------------------------------
// Tensor Data Mover: 2-D tile load Global -> LDS (bf16 elements).
// D# layout per CDNA5 ISA (groups 0/1; groups 2/3 = 0
// for a 2-D tensor). This toolchain exposes the 6-arg builtin
// (g0 u32x4, g1 i32x8, g2 i32x4, g3 i32x4, g4 i32x8, cpol i32).
// Issued once per wave (EXEC ignored) -> call from wave 0 only.
//   tile_x    : tile width  (elements, contiguous dim; *2B must be mult of 4)
//   tile_y    : tile height (rows)
//   stride_x  : tensor row stride (elements)
// ---------------------------------------------------------------------------
DEV void tdm_load_2d_bf16(unsigned lds_byte_off, const void* gaddr,
                          unsigned tile_x, unsigned tile_y,
                          unsigned tensor_x, unsigned tensor_y,
                          unsigned long long stride_x) {
  const unsigned long long ga = (unsigned long long)gaddr;
  u32x4 g0;
  g0[0] = 1u;                                        // count=1, user mode
  g0[1] = lds_byte_off;                              // lds_addr (bytes)
  g0[2] = (unsigned)(ga & 0xffffffffull);            // global_addr[31:0]
  g0[3] = (unsigned)((ga >> 32) & 0x1ffffffull)      // global_addr[56:32]
          | (2u << 30);                              // type = 2 ("image")
  i32x8 g1;
  g1[0] = (int)(1u << 16);                           // wg_mask=0, data_size=1 (2B)
  g1[1] = (int)((tensor_x & 0xffffu) << 16);         // tensor_dim0[15:0]
  g1[2] = (int)((tensor_x >> 16) |                   // tensor_dim0[31:16]
                ((tensor_y & 0xffffu) << 16));       // tensor_dim1[15:0]
  g1[3] = (int)((tensor_y >> 16) |                   // tensor_dim1[31:16]
                ((tile_x & 0xffffu) << 16));         // tile_dim0
  g1[4] = (int)(tile_y & 0xffffu);                   // tile_dim1 (tile_dim2=0)
  g1[5] = (int)(stride_x & 0xffffffffull);           // tensor_dim0_stride[31:0]
  g1[6] = (int)((stride_x >> 32) & 0xffffull);       // stride[47:32]; dim1_stride=0
  g1[7] = 0;
  const i32x4 z4 = {0, 0, 0, 0};
  const i32x8 z8 = {0, 0, 0, 0, 0, 0, 0, 0};
  __builtin_amdgcn_tensor_load_to_lds(g0, g1, z4, z4, z8, 0);
}

DEV unsigned lds_offset_of(const void* p) {
  // generic LDS pointers carry the group-segment offset in the low 32 bits
  return (unsigned)(unsigned long long)p;
}

// ---------------------------------------------------------------------------
// Elementwise cast f32 -> bf16 (vector x4)
// ---------------------------------------------------------------------------
__global__ void __launch_bounds__(256)
cast4_f32_bf16(const float* __restrict__ in, __bf16* __restrict__ out, int n4) {
  int i = blockIdx.x * 256 + threadIdx.x;
  if (i < n4) {
    float4 f = ((const float4*)in)[i];
    __bf16* o = out + (size_t)i * 4;
    o[0] = (__bf16)f.x; o[1] = (__bf16)f.y;
    o[2] = (__bf16)f.z; o[3] = (__bf16)f.w;
  }
}

// Cast + transpose: in [rows, cols] f32 -> out [cols, rows] bf16
__global__ void __launch_bounds__(256)
castT_f32_bf16(const float* __restrict__ in, __bf16* __restrict__ out,
               int rows, int cols) {
  int c = blockIdx.x * 256 + threadIdx.x;
  int r = blockIdx.y;
  if (c < cols) out[(size_t)c * rows + r] = (__bf16)in[(size_t)r * cols + c];
}

// ---------------------------------------------------------------------------
// Small-N WMMA GEMM (direct from global): C = A @ Bt^T + bias
// 4 waves/block, one 16x16 tile per wave. Used for K/V projections (N=64).
// ---------------------------------------------------------------------------
template<bool STORE_TRANS, bool OUT_F32>
__global__ void __launch_bounds__(128)
wmma_gemm_bias(const __bf16* __restrict__ A, const __bf16* __restrict__ Bt,
               const float* __restrict__ bias, void* __restrict__ Cout,
               int M, int N, int K, int ldc) {
  const int lane = threadIdx.x & 31;
  const int wv   = threadIdx.x >> 5;
  const int lr   = lane & 15;
  const int hiw  = lane >> 4;          // 0: lanes 0-15, 1: lanes 16-31
  const int n0   = (blockIdx.x * 4 + wv) * 16;
  const int m0   = blockIdx.y * 16;
  if (n0 >= N || m0 >= M) return;

  const __bf16* arow = A  + (size_t)(m0 + lr) * K;   // A: row = lane%16
  const __bf16* brow = Bt + (size_t)(n0 + lr) * K;   // B: col = lane%16
  const int aoff = hiw * 16;   // A 16x32: upper half-wave -> K+16..K+31
  const int boff = hiw * 16;   // B 32x16: upper half-wave -> K+16..K+31

  v8f c = {};
  for (int kk = 0; kk < K; kk += 32) {
    v16bf a = load_frag(arow + kk + aoff, arow + kk + aoff + 8);
    v16bf b = load_frag(brow + kk + boff, brow + kk + boff + 8);
    if (kk + 32 < K) {                       // global_prefetch_b8 next tiles
      __builtin_prefetch(arow + kk + 32 + aoff, 0, 3);
      __builtin_prefetch(brow + kk + 32 + boff, 0, 3);
    }
    c = wmma_bf16(a, b, c);
  }

  const float bv = bias[n0 + lr];
#pragma unroll
  for (int r = 0; r < 8; ++r) {
    const int row = m0 + r + hiw * 8;        // C layout: VGPR r -> rows r/r+8
    const int col = n0 + lr;
    const float val = c[r] + bv;
    if (OUT_F32) {
      ((float*)Cout)[(size_t)row * ldc + col] = val;
    } else if (STORE_TRANS) {
      ((__bf16*)Cout)[(size_t)col * ldc + row] = (__bf16)val;
    } else {
      ((__bf16*)Cout)[(size_t)row * ldc + col] = (__bf16)val;
    }
  }
}

// ---------------------------------------------------------------------------
// Big WMMA GEMM: TDM (tensor_load_to_lds) double-buffered LDS tiling.
// Block = 256 threads (8 waves), tile 128(M) x 128(N), K-step 32.
// Wave w owns rows [m0+16w, m0+16w+16) across all 128 N columns (8 acc tiles).
// TDM issued by wave 0 only; completion via s_wait_tensorcnt before barrier.
// ---------------------------------------------------------------------------
template<bool OUT_F32>
__global__ void __launch_bounds__(256)
wmma_gemm_lds(const __bf16* __restrict__ A, const __bf16* __restrict__ Bt,
              const float* __restrict__ bias, void* __restrict__ Cout,
              int M, int N, int K, int ldc) {
  __shared__ __align__(16) __bf16 ldsA[2][128 * 32];   // 8 KB x2
  __shared__ __align__(16) __bf16 ldsB[2][128 * 32];   // 8 KB x2

  const int lane = threadIdx.x & 31;
  const int w    = threadIdx.x >> 5;       // 0..7
  const int lr   = lane & 15;
  const int hiw  = lane >> 4;
  const int m0   = blockIdx.y * 128;
  const int n0   = blockIdx.x * 128;
  const int aoff = hiw * 16;
  const int boff = hiw * 16;
  const bool issuer = (threadIdx.x < 32);  // wave 0 drives the TDM

  const int T = K / 32;
  v8f acc[8] = {{}, {}, {}, {}, {}, {}, {}, {}};

  if (issuer) {
    tdm_load_2d_bf16(lds_offset_of(&ldsA[0][0]), A + (size_t)m0 * K,
                     32, 128, (unsigned)K, (unsigned)M, (unsigned)K);
    tdm_load_2d_bf16(lds_offset_of(&ldsB[0][0]), Bt + (size_t)n0 * K,
                     32, 128, (unsigned)K, (unsigned)N, (unsigned)K);
  }

  for (int i = 0; i < T; ++i) {
    const int cur = i & 1;
    if (issuer) {
      if (i + 1 < T) {
        const int nxt = cur ^ 1;
        tdm_load_2d_bf16(lds_offset_of(&ldsA[nxt][0]),
                         A + (size_t)m0 * K + (size_t)(i + 1) * 32,
                         32, 128, (unsigned)K, (unsigned)M, (unsigned)K);
        tdm_load_2d_bf16(lds_offset_of(&ldsB[nxt][0]),
                         Bt + (size_t)n0 * K + (size_t)(i + 1) * 32,
                         32, 128, (unsigned)K, (unsigned)N, (unsigned)K);
        __builtin_amdgcn_s_wait_tensorcnt(2);   // tile i complete (in-order)
      } else {
        __builtin_amdgcn_s_wait_tensorcnt(0);   // final tile complete
      }
    }
    __syncthreads();                            // LDS tile i visible to all

    const __bf16* ar = &ldsA[cur][(w * 16 + lr) * 32];
    v16bf a = load_frag(ar + aoff, ar + aoff + 8);
#pragma unroll
    for (int t = 0; t < 8; ++t) {
      const __bf16* br = &ldsB[cur][(t * 16 + lr) * 32];
      v16bf b = load_frag(br + boff, br + boff + 8);
      acc[t] = wmma_bf16(a, b, acc[t]);
    }
    __syncthreads();                            // done reading before overwrite
  }

#pragma unroll
  for (int t = 0; t < 8; ++t) {
    const int col = n0 + t * 16 + lr;
    const float bv = bias[col];
#pragma unroll
    for (int r = 0; r < 8; ++r) {
      const int row = m0 + w * 16 + r + hiw * 8;
      const float val = acc[t][r] + bv;
      if (OUT_F32) ((float*)Cout)[(size_t)row * ldc + col] = val;
      else         ((__bf16*)Cout)[(size_t)row * ldc + col] = (__bf16)val;
    }
  }
}

// ---------------------------------------------------------------------------
// Flash MQA attention, one wave per (b, h, 16-query tile).
//  Q  : [B*S, 1024] bf16 (head h at column h*64)
//  Kh : [B*S, 64]   bf16 (shared KV head)
//  Vt : [64, B*S]   bf16 (V transposed)
//  AO : [B*S, 1024] bf16 (heads concatenated)
// ---------------------------------------------------------------------------
#define S_LEN 2048
#define DMOD  1024

__global__ void __launch_bounds__(32)
mqa_flash(const __bf16* __restrict__ Q, const __bf16* __restrict__ Kh,
          const __bf16* __restrict__ Vt, __bf16* __restrict__ AO) {
  __shared__ __align__(16) __bf16 plds[16 * 32];   // P tile staging (C->A layout)

  const int lane = threadIdx.x & 31;
  const int lr   = lane & 15;
  const int hiw  = lane >> 4;
  const int b    = blockIdx.z;
  const int h    = blockIdx.y;
  const int q0   = blockIdx.x * 16;

  const int aoff = hiw * 16;
  const int boff = hiw * 16;

  // Q fragments: 16x64 held as two 16x32 A-fragments, resident all loop.
  const __bf16* qrow = Q + (size_t)(b * S_LEN + q0 + lr) * DMOD + h * 64;
  const v16bf qa0 = load_frag(qrow + aoff,      qrow + aoff + 8);
  const v16bf qa1 = load_frag(qrow + 32 + aoff, qrow + 32 + aoff + 8);

  const __bf16* kbase = Kh + (size_t)(b * S_LEN) * 64;
  const size_t  vstride = (size_t)4 * S_LEN;       // Vt leading dim = B*S
  const size_t  vcol0   = (size_t)b * S_LEN;

  float m[8], l[8];
  v8f o[4] = {{}, {}, {}, {}};
#pragma unroll
  for (int r = 0; r < 8; ++r) { m[r] = -3.0e38f; l[r] = 0.0f; }

  for (int kt = 0; kt < S_LEN; kt += 32) {
    // ---- scores: two 16x16 tiles (keys kt..+15, kt+16..+31), K-dim = dk=64
    v8f c0 = {}, c1 = {};
    {
      const __bf16* kr0 = kbase + (size_t)(kt + lr) * 64;
      v16bf b00 = load_frag(kr0 + boff,      kr0 + boff + 8);       // dk 0..31
      v16bf b01 = load_frag(kr0 + 32 + boff, kr0 + 32 + boff + 8);  // dk 32..63
      c0 = wmma_bf16(qa0, b00, c0);
      c0 = wmma_bf16(qa1, b01, c0);
      const __bf16* kr1 = kbase + (size_t)(kt + 16 + lr) * 64;
      v16bf b10 = load_frag(kr1 + boff,      kr1 + boff + 8);
      v16bf b11 = load_frag(kr1 + 32 + boff, kr1 + 32 + boff + 8);
      c1 = wmma_bf16(qa0, b10, c1);
      c1 = wmma_bf16(qa1, b11, c1);
    }

    // ---- online softmax over the 32-key slab
#pragma unroll
    for (int r = 0; r < 8; ++r) {
      float s0 = c0[r] * 0.125f;               // 1/sqrt(dk)
      float s1 = c1[r] * 0.125f;
      float mx = fmaxf(s0, s1);                // row lives in one half-wave:
      mx = fmaxf(mx, __shfl_xor(mx, 1));       // masks 1..8 stay in-half
      mx = fmaxf(mx, __shfl_xor(mx, 2));
      mx = fmaxf(mx, __shfl_xor(mx, 4));
      mx = fmaxf(mx, __shfl_xor(mx, 8));
      const float mnew = fmaxf(m[r], mx);
      const float sc   = __expf(m[r] - mnew);
      m[r] = mnew;
      const float p0 = __expf(s0 - mnew);
      const float p1 = __expf(s1 - mnew);
      float ps = p0 + p1;
      ps += __shfl_xor(ps, 1);
      ps += __shfl_xor(ps, 2);
      ps += __shfl_xor(ps, 4);
      ps += __shfl_xor(ps, 8);
      l[r] = l[r] * sc + ps;
      o[0][r] *= sc; o[1][r] *= sc; o[2][r] *= sc; o[3][r] *= sc;
      const int prow = r + hiw * 8;            // C layout -> LDS row-major
      plds[prow * 32 + lr]      = (__bf16)p0;
      plds[prow * 32 + 16 + lr] = (__bf16)p1;
    }

    // single wave: DS ops are in-order; explicit wait keeps loads after stores
    asm volatile("s_wait_dscnt 0" ::: "memory");

    // ---- re-read P as a 16x32 A-fragment
    const __bf16* pr = &plds[lr * 32 + aoff];
    const v16bf pa = load_frag(pr, pr + 8);

    // ---- PV: out[16x64] += P[16x32] @ V[32x64]
#pragma unroll
    for (int n = 0; n < 4; ++n) {
      const __bf16* vr = Vt + (size_t)(n * 16 + lr) * vstride + vcol0 + kt;
      v16bf vb = load_frag(vr + boff, vr + boff + 8);
      o[n] = wmma_bf16(pa, vb, o[n]);
    }
  }

  // ---- finalize: divide by softmax denom, store bf16 head slice
#pragma unroll
  for (int r = 0; r < 8; ++r) {
    const float inv = 1.0f / l[r];
    const int row = q0 + r + hiw * 8;
    __bf16* orow = AO + (size_t)(b * S_LEN + row) * DMOD + h * 64 + lr;
    orow[0]  = (__bf16)(o[0][r] * inv);
    orow[16] = (__bf16)(o[1][r] * inv);
    orow[32] = (__bf16)(o[2][r] * inv);
    orow[48] = (__bf16)(o[3][r] * inv);
  }
}

// ---------------------------------------------------------------------------
// Host launcher
// ---------------------------------------------------------------------------
extern "C" void kernel_launch(void* const* d_in, const int* in_sizes, int n_in,
                              void* d_out, int out_size, void* d_ws, size_t ws_size,
                              hipStream_t stream) {
  (void)in_sizes; (void)n_in; (void)out_size; (void)ws_size;

  const float* q  = (const float*)d_in[0];
  const float* k  = (const float*)d_in[1];
  const float* v  = (const float*)d_in[2];
  const float* Wq = (const float*)d_in[3];
  const float* bq = (const float*)d_in[4];
  const float* Wk = (const float*)d_in[5];
  const float* bk = (const float*)d_in[6];
  const float* Wv = (const float*)d_in[7];
  const float* bv = (const float*)d_in[8];
  const float* Wo = (const float*)d_in[9];
  const float* bo = (const float*)d_in[10];
  float* out = (float*)d_out;

  const int B = 4, S = S_LEN, DM = DMOD, DK = 64;
  const size_t BS = (size_t)B * S;                 // 8192

  // workspace carve-up (~87 MB), 256B aligned
  char* ws = (char*)d_ws;
  size_t off = 0;
  auto carve = [&](size_t bytes) -> void* {
    void* p = ws + off;
    off += (bytes + 255) & ~(size_t)255;
    return p;
  };
  __bf16* qb  = (__bf16*)carve(BS * DM * 2);
  __bf16* kb  = (__bf16*)carve(BS * DM * 2);
  __bf16* vb  = (__bf16*)carve(BS * DM * 2);
  __bf16* WqT = (__bf16*)carve((size_t)DM * DM * 2);
  __bf16* WkT = (__bf16*)carve((size_t)DK * DM * 2);
  __bf16* WvT = (__bf16*)carve((size_t)DK * DM * 2);
  __bf16* WoT = (__bf16*)carve((size_t)DM * DM * 2);
  __bf16* Qh  = (__bf16*)carve(BS * DM * 2);
  __bf16* Kh  = (__bf16*)carve(BS * DK * 2);
  __bf16* Vt  = (__bf16*)carve(BS * DK * 2);
  __bf16* AO  = (__bf16*)carve(BS * DM * 2);

  // 1) casts: activations (flat), weights (transposed to [out,in])
  const int n4 = (int)(BS * DM / 4);
  cast4_f32_bf16<<<(n4 + 255) / 256, 256, 0, stream>>>(q, qb, n4);
  cast4_f32_bf16<<<(n4 + 255) / 256, 256, 0, stream>>>(k, kb, n4);
  cast4_f32_bf16<<<(n4 + 255) / 256, 256, 0, stream>>>(v, vb, n4);
  castT_f32_bf16<<<dim3((DM + 255) / 256, DM), 256, 0, stream>>>(Wq, WqT, DM, DM);
  castT_f32_bf16<<<dim3(1, DM),                256, 0, stream>>>(Wk, WkT, DM, DK);
  castT_f32_bf16<<<dim3(1, DM),                256, 0, stream>>>(Wv, WvT, DM, DK);
  castT_f32_bf16<<<dim3((DM + 255) / 256, DM), 256, 0, stream>>>(Wo, WoT, DM, DM);

  // 2) projections (WMMA). Big GEMMs take the TDM/LDS-tiled path.
  wmma_gemm_lds<false><<<dim3(DM / 128, (int)(BS / 128)), 256, 0, stream>>>(
      qb, WqT, bq, Qh, (int)BS, DM, DM, DM);                 // Qh [8192,1024] bf16
  wmma_gemm_bias<false, false><<<dim3(DK / 64, (int)(BS / 16)), 128, 0, stream>>>(
      kb, WkT, bk, Kh, (int)BS, DK, DM, DK);                 // Kh [8192,64] bf16
  wmma_gemm_bias<true, false><<<dim3(DK / 64, (int)(BS / 16)), 128, 0, stream>>>(
      vb, WvT, bv, Vt, (int)BS, DK, DM, (int)BS);            // Vt [64,8192] bf16

  // 3) flash MQA attention (WMMA scores + WMMA PV, online softmax)
  mqa_flash<<<dim3(S / 16, 16, B), 32, 0, stream>>>(Qh, Kh, Vt, AO);

  // 4) output projection -> f32 d_out (TDM/LDS-tiled)
  wmma_gemm_lds<true><<<dim3(DM / 128, (int)(BS / 128)), 256, 0, stream>>>(
      AO, WoT, bo, out, (int)BS, DM, DM, DM);
}
